// GruTransitionModel_28647431864358
// MI455X (gfx1250) — matrix operand order repaired and checked by reading, not compiled, not run on hardware
//
#include <hip/hip_runtime.h>
#include <hip/hip_bf16.h>
#include <stdint.h>

// ---------------- problem constants ----------------
#define EMB   200
#define SDIM  12
#define ADIM  4
#define CDIM  8
#define LCDIM 6
#define TSTEPS 100
#define BSZ   2048
#define NT    (TSTEPS * BSZ)        // 204800 rows (per-timestep batch)
#define NH    ((TSTEPS + 1) * BSZ)  // 206848 rows (hiddens incl. h0)
#define KP_H  224                   // EMB padded to 7*32 (WMMA K granularity)
#define KP_E  416                   // 2*EMB padded to 13*32
#define MP_E  208                   // 200 padded to 16 multiple
#define MP_G  608                   // 600 padded to 16 multiple
#define LD_G  608
#define CB    4                     // column tiles (16 cols each) per wave

// ---------------- types for WMMA ----------------
typedef __attribute__((ext_vector_type(16))) __bf16        v16bf;
typedef __attribute__((ext_vector_type(8)))  float         v8f;
typedef __attribute__((ext_vector_type(4)))  unsigned int  u32x4;

union ABfrag { u32x4 u[2]; v16bf v; };

__device__ __forceinline__ unsigned short f2bf(float f) {
  unsigned u = __float_as_uint(f);
  u += 0x7FFFu + ((u >> 16) & 1u);   // round-to-nearest-even
  return (unsigned short)(u >> 16);
}

// ---------------- generic bf16 WMMA GEMM, 16x64 tile per wave ----------------
// C[N, M] = act( A[N, KP](bf16) @ W[Mp, KP](bf16)^T + bias[col] + addRow[(row&mask), col] )
// One wave computes 4 adjacent 16x16 tiles. Per K-step: all 10 fragment loads are
// issued first (distinct live ranges -> staged s_wait_loadcnt), then 4 WMMAs.
// Fragment addressing follows the CDNA5 ISA 16-bit layouts:
//   A (16x32): lane l<16 -> row l, K {k0..k0+7, k0+16..k0+23}; lane l>=16 -> row l-16, K {+8, +24}
//   B (32x16): lane n<16 -> col n, K {k0..k0+7, k0+16..k0+23}; lane n>=16 -> col n-16, K {+8, +24}
__global__ __launch_bounds__(32) void gemm_bf16_wmma(
    const unsigned short* __restrict__ A, int ldA,
    const unsigned short* __restrict__ W, int KP, int MpT,   // MpT = Mp/16 column tiles
    const float* __restrict__ bias,
    float* __restrict__ outF, int ldF,
    unsigned short* __restrict__ outB, int ldB, int colOffB,
    const float* __restrict__ addRow, int addLd, int rowMask,
    int Mreal, int act)
{
  const int it = blockIdx.x;                 // tile over N (rows)
  const int jb = blockIdx.y * CB;            // first column tile of this wave
  const int i0 = it << 4;
  const int lane = threadIdx.x;
  const int l  = lane & 15;
  const int hi = lane >> 4;

  const unsigned short* aptr = A + (size_t)(i0 + l) * (size_t)ldA + hi * 8;

  int ct[CB];
  const unsigned short* wptr[CB];
#pragma unroll
  for (int s = 0; s < CB; ++s) {
    int c = jb + s;
    if (c >= MpT) c = MpT - 1;               // clamp: duplicate compute, dedup'd at store
    ct[s] = c;
    wptr[s] = W + (size_t)((c << 4) + l) * (size_t)KP + hi * 8;
  }

  const v8f vzero = {0.f, 0.f, 0.f, 0.f, 0.f, 0.f, 0.f, 0.f};
  v8f acc[CB];
#pragma unroll
  for (int s = 0; s < CB; ++s) acc[s] = vzero;

  for (int k0 = 0; k0 < KP; k0 += 32) {
    ABfrag a, b[CB];
    // Issue every load for this K-step first: 10 b128 loads with independent
    // destinations, so the WMMAs below can use staged (non-zero) load waits.
    a.u[0] = *(const u32x4*)(aptr + k0);        // 8 bf16 (K k0..k0+7 / +8..+15)
    a.u[1] = *(const u32x4*)(aptr + k0 + 16);   // 8 bf16 (K +16..+23 / +24..+31)
#pragma unroll
    for (int s = 0; s < CB; ++s) {
      b[s].u[0] = *(const u32x4*)(wptr[s] + k0);     // 8 bf16 (K k0..k0+7 / +8..+15)
      b[s].u[1] = *(const u32x4*)(wptr[s] + k0 + 16);
    }
#pragma unroll
    for (int s = 0; s < CB; ++s) {
      acc[s] = __builtin_amdgcn_wmma_f32_16x16x32_bf16(false, a.v, false, b[s].v,
                                                       (short)0, acc[s], false, false);
    }
  }

  // C/D layout: lane<16 -> N=lane, rows i0..i0+7 in elems 0..7; lane>=16 -> rows i0+8..i0+15
  const int rbase = i0 + hi * 8;
#pragma unroll
  for (int s = 0; s < CB; ++s) {
    if (s > 0 && (jb + s) >= MpT) continue;    // clamped duplicate tile -> skip store
    const int col = (ct[s] << 4) + l;
    if (col < Mreal) {
      const float bv = bias ? bias[col] : 0.f;
#pragma unroll
      for (int v = 0; v < 8; ++v) {
        const int row = rbase + v;
        float x = acc[s][v] + bv;
        if (addRow) x += addRow[(size_t)(row & rowMask) * (size_t)addLd + col];
        if (act == 1)      x = fmaxf(x, 0.f);
        else if (act == 2) x = tanhf(x);
        else if (act == 3) {                   // sqrt(softplus(x) + eps)
          float sp = (x > 20.f) ? x : log1pf(__expf(x));
          x = sqrtf(sp + 1e-4f);
        }
        if (outF) outF[(size_t)row * (size_t)ldF + col] = x;
        if (outB) outB[(size_t)row * (size_t)ldB + colOffB + col] = f2bf(x);
      }
    }
  }
}

// ---------------- skinny-K layer 1 (K in {4,6,8,12}): VALU, writes K-padded bf16 ----------------
__global__ void layer1_relu_pad(const float* __restrict__ x, int Din,
                                const float* __restrict__ w1, const float* __restrict__ b1,
                                unsigned short* __restrict__ out, long N)
{
  long idx = (long)blockIdx.x * blockDim.x + threadIdx.x;
  long total = N * (long)KP_H;
  if (idx >= total) return;
  long n = idx / KP_H;
  int  j = (int)(idx - n * KP_H);
  float acc = 0.f;
  if (j < EMB) {
    acc = b1[j];
    const float* xr = x  + n * (long)Din;
    const float* wr = w1 + (long)j * Din;
    for (int k = 0; k < Din; ++k) acc = fmaf(xr[k], wr[k], acc);
    acc = fmaxf(acc, 0.f);
  }
  out[idx] = f2bf(acc);
}

// ---------------- weight repack fp32 -> zero-padded bf16 [Mp, KP] ----------------
__global__ void pack_w(const float* __restrict__ src, unsigned short* __restrict__ dst,
                       int Mp, int Msrc, int KP, int Kcopy, int srcLd, int srcOff)
{
  int idx = blockIdx.x * blockDim.x + threadIdx.x;
  if (idx >= Mp * KP) return;
  int m = idx / KP, k = idx - m * KP;
  float v = (m < Msrc && k < Kcopy) ? src[(long)m * srcLd + srcOff + k] : 0.f;
  dst[idx] = f2bf(v);
}

// ---------------- zero pad columns [c0, ld) of a bf16 buffer ----------------
__global__ void zero_cols(unsigned short* __restrict__ buf, long rows, int ld, int c0)
{
  int pc = ld - c0;
  long idx = (long)blockIdx.x * blockDim.x + threadIdx.x;
  long total = rows * (long)pc;
  if (idx >= total) return;
  long r = idx / pc;
  int  c = c0 + (int)(idx - r * pc);
  buf[r * (long)ld + c] = 0;
}

// ---------------- GRU gate fusion (elementwise, per step) ----------------
__global__ void gru_gates_kernel(const float* __restrict__ gi, const float* __restrict__ gh,
                                 float* __restrict__ h, unsigned short* __restrict__ hidNext)
{
  int idx = blockIdx.x * blockDim.x + threadIdx.x;
  if (idx >= BSZ * EMB) return;
  int b = idx / EMB, j = idx - b * EMB;
  const float* gib = gi + (size_t)b * LD_G;
  const float* ghb = gh + (size_t)b * LD_G;
  float r = 1.f / (1.f + __expf(-(gib[j] + ghb[j])));
  float z = 1.f / (1.f + __expf(-(gib[j + EMB] + ghb[j + EMB])));
  float n = tanhf(gib[j + 2 * EMB] + r * ghb[j + 2 * EMB]);
  float hv = h[idx];
  float hn = (1.f - z) * n + z * hv;
  h[idx] = hn;
  hidNext[(size_t)b * KP_H + j] = f2bf(hn);
}

// ---------------- host orchestration ----------------
extern "C" void kernel_launch(void* const* d_in, const int* in_sizes, int n_in,
                              void* d_out, int out_size, void* d_ws, size_t ws_size,
                              hipStream_t stream)
{
  (void)in_sizes; (void)n_in; (void)out_size; (void)ws_size;
  const float* x         = (const float*)d_in[0];
  const float* u         = (const float*)d_in[1];
  const float* ctx_mean  = (const float*)d_in[2];
  const float* local_ctx = (const float*)d_in[3];
  const float* se_w1 = (const float*)d_in[4],  *se_b1 = (const float*)d_in[5];
  const float* se_w2 = (const float*)d_in[6],  *se_b2 = (const float*)d_in[7];
  const float* ae_w1 = (const float*)d_in[8],  *ae_b1 = (const float*)d_in[9];
  const float* ae_w2 = (const float*)d_in[10], *ae_b2 = (const float*)d_in[11];
  const float* ce_w1 = (const float*)d_in[12], *ce_b1 = (const float*)d_in[13];
  const float* ce_w2 = (const float*)d_in[14], *ce_b2 = (const float*)d_in[15];
  const float* lce_w1= (const float*)d_in[16], *lce_b1= (const float*)d_in[17];
  const float* lce_w2= (const float*)d_in[18], *lce_b2= (const float*)d_in[19];
  const float* dec_w1= (const float*)d_in[20], *dec_b1= (const float*)d_in[21];
  const float* dec_w2= (const float*)d_in[22], *dec_b2= (const float*)d_in[23];
  const float* gru_wih=(const float*)d_in[24], *gru_bih=(const float*)d_in[25];
  const float* gru_whh=(const float*)d_in[26], *gru_bhh=(const float*)d_in[27];
  const float* cov_w1= (const float*)d_in[28], *cov_b1= (const float*)d_in[29];
  const float* cov_w2= (const float*)d_in[30], *cov_b2= (const float*)d_in[31];
  float* out = (float*)d_out;

  // ---- workspace carve (256B aligned) ----
  char* base = (char*)d_ws;
  size_t off = 0;
  auto take = [&](size_t bytes) -> char* {
    char* p = base + off;
    off = (off + bytes + 255) & ~(size_t)255;
    return p;
  };
  unsigned short* E    = (unsigned short*)take((size_t)NT  * KP_E * 2); // [a_emb | lce] per (t,b)
  unsigned short* H1   = (unsigned short*)take((size_t)NH  * KP_H * 2); // layer-1 temp (reused)
  unsigned short* HID  = (unsigned short*)take((size_t)NH  * KP_H * 2); // hiddens (bf16, K-padded)
  float* GCB = (float*)take((size_t)BSZ * LD_G * 4);  // b_ih + ctx_embed @ Wih[:, :200]^T
  float* GI  = (float*)take((size_t)BSZ * LD_G * 4);  // per-step gi
  float* GH  = (float*)take((size_t)BSZ * LD_G * 4);  // per-step gh
  float* HF  = (float*)take((size_t)BSZ * EMB  * 4);  // fp32 recurrent state
  unsigned short* CTXE = (unsigned short*)take((size_t)BSZ * KP_H * 2);
  unsigned short* Wse2 = (unsigned short*)take((size_t)MP_E * KP_H * 2);
  unsigned short* Wae2 = (unsigned short*)take((size_t)MP_E * KP_H * 2);
  unsigned short* Wce2 = (unsigned short*)take((size_t)MP_E * KP_H * 2);
  unsigned short* Wlce2= (unsigned short*)take((size_t)MP_E * KP_H * 2);
  unsigned short* Wdec1= (unsigned short*)take((size_t)MP_E * KP_H * 2);
  unsigned short* Wcov1= (unsigned short*)take((size_t)MP_E * KP_H * 2);
  unsigned short* Wdec2= (unsigned short*)take((size_t)16   * KP_H * 2);
  unsigned short* Wcov2= (unsigned short*)take((size_t)16   * KP_H * 2);
  unsigned short* Wc   = (unsigned short*)take((size_t)MP_G * KP_H * 2); // Wih[:, 0:200]
  unsigned short* Whh  = (unsigned short*)take((size_t)MP_G * KP_H * 2);
  unsigned short* Wal  = (unsigned short*)take((size_t)MP_G * KP_E * 2); // Wih[:, 200:600]

  auto pack = [&](const float* src, unsigned short* dst, int Mp, int Msrc,
                  int KP, int Kcopy, int srcLd, int srcOff) {
    int total = Mp * KP;
    pack_w<<<dim3((total + 255) / 256), dim3(256), 0, stream>>>(src, dst, Mp, Msrc, KP, Kcopy, srcLd, srcOff);
  };
  auto gemm = [&](const unsigned short* A, int ldA, const unsigned short* W, int KP,
                  const float* bias, float* outF, int ldF,
                  unsigned short* outB, int ldB, int colOffB,
                  const float* addRow, int addLd, int rowMask,
                  int N, int Mp, int Mreal, int act) {
    int MpT = Mp / 16;
    gemm_bf16_wmma<<<dim3(N / 16, (MpT + CB - 1) / CB), dim3(32), 0, stream>>>(
        A, ldA, W, KP, MpT, bias, outF, ldF, outB, ldB, colOffB,
        addRow, addLd, rowMask, Mreal, act);
  };
  auto l1 = [&](const float* xin, int Din, const float* w1, const float* b1,
                unsigned short* o, long N) {
    long total = N * (long)KP_H;
    layer1_relu_pad<<<dim3((unsigned)((total + 255) / 256)), dim3(256), 0, stream>>>(xin, Din, w1, b1, o, N);
  };
  auto zpad = [&](unsigned short* buf, long rows, int ld, int c0) {
    long total = rows * (long)(ld - c0);
    zero_cols<<<dim3((unsigned)((total + 255) / 256)), dim3(256), 0, stream>>>(buf, rows, ld, c0);
  };

  // ---- weight repacks (bf16, zero padded) ----
  pack(se_w2,  Wse2,  MP_E, EMB, KP_H, EMB, EMB, 0);
  pack(ae_w2,  Wae2,  MP_E, EMB, KP_H, EMB, EMB, 0);
  pack(ce_w2,  Wce2,  MP_E, EMB, KP_H, EMB, EMB, 0);
  pack(lce_w2, Wlce2, MP_E, EMB, KP_H, EMB, EMB, 0);
  pack(dec_w1, Wdec1, MP_E, EMB, KP_H, EMB, EMB, 0);
  pack(cov_w1, Wcov1, MP_E, EMB, KP_H, EMB, EMB, 0);
  pack(dec_w2, Wdec2, 16, SDIM, KP_H, EMB, EMB, 0);
  pack(cov_w2, Wcov2, 16, SDIM, KP_H, EMB, EMB, 0);
  pack(gru_wih, Wc,  MP_G, 600, KP_H, EMB,     600, 0);    // ctx slice
  pack(gru_wih, Wal, MP_G, 600, KP_E, 2 * EMB, 600, EMB);  // [a|lc] slice
  pack(gru_whh, Whh, MP_G, 600, KP_H, EMB,     EMB, 0);

  // ---- zero pad columns of streamed bf16 activation buffers ----
  zpad(E,    NT,  KP_E, 2 * EMB);
  zpad(HID,  NH,  KP_H, EMB);
  zpad(H1,   NH,  KP_H, EMB);
  zpad(CTXE, BSZ, KP_H, EMB);

  // ---- embeddings ----
  // state_embed -> h0 (fp32 HF) and HID rows [0, BSZ)
  l1(x, SDIM, se_w1, se_b1, H1, BSZ);
  gemm(H1, KP_H, Wse2, KP_H, se_b2, HF, EMB, HID, KP_H, 0,
       nullptr, 0, 0, BSZ, MP_E, EMB, /*tanh*/2);
  // ctx_embed
  l1(ctx_mean, CDIM, ce_w1, ce_b1, H1, BSZ);
  gemm(H1, KP_H, Wce2, KP_H, ce_b2, nullptr, 0, CTXE, KP_H, 0,
       nullptr, 0, 0, BSZ, MP_E, EMB, /*relu*/1);
  // action embeddings for all T -> E[:, 0:200]
  l1(u, ADIM, ae_w1, ae_b1, H1, NT);
  gemm(H1, KP_H, Wae2, KP_H, ae_b2, nullptr, 0, E, KP_E, 0,
       nullptr, 0, 0, NT, MP_E, EMB, 1);
  // local-context embeddings for all T -> E[:, 200:400]
  l1(local_ctx, LCDIM, lce_w1, lce_b1, H1, NT);
  gemm(H1, KP_H, Wlce2, KP_H, lce_b2, nullptr, 0, E, KP_E, EMB,
       nullptr, 0, 0, NT, MP_E, EMB, 1);
  // GCB = b_ih + ctx_embed @ Wih[:, :200]^T   (constant across t)
  gemm(CTXE, KP_H, Wc, KP_H, gru_bih, GCB, LD_G, nullptr, 0, 0,
       nullptr, 0, 0, BSZ, MP_G, 600, 0);

  // ---- sequential GRU scan ----
  for (int t = 0; t < TSTEPS; ++t) {
    // gh = h_t @ Whh^T + b_hh
    gemm(HID + (size_t)t * BSZ * KP_H, KP_H, Whh, KP_H, gru_bhh,
         GH, LD_G, nullptr, 0, 0, nullptr, 0, 0, BSZ, MP_G, 600, 0);
    // gi = [a_t | lc_t] @ Wih[:,200:600]^T + GCB
    gemm(E + (size_t)t * BSZ * KP_E, KP_E, Wal, KP_E, nullptr,
         GI, LD_G, nullptr, 0, 0, GCB, LD_G, BSZ - 1, BSZ, MP_G, 600, 0);
    // gates -> h_{t+1} (fp32 in HF, bf16 into HID row block t+1)
    gru_gates_kernel<<<dim3((BSZ * EMB + 255) / 256), dim3(256), 0, stream>>>(
        GI, GH, HF, HID + (size_t)(t + 1) * BSZ * KP_H);
  }

  // ---- decode heads over all T+1 hiddens ----
  // state_mean
  gemm(HID, KP_H, Wdec1, KP_H, dec_b1, nullptr, 0, H1, KP_H, 0,
       nullptr, 0, 0, NH, MP_E, EMB, 1);
  gemm(H1, KP_H, Wdec2, KP_H, dec_b2, out, SDIM, nullptr, 0, 0,
       nullptr, 0, 0, NH, 16, SDIM, 0);
  // state_std = sqrt(softplus(.) + eps)
  gemm(HID, KP_H, Wcov1, KP_H, cov_b1, nullptr, 0, H1, KP_H, 0,
       nullptr, 0, 0, NH, MP_E, EMB, 1);
  gemm(H1, KP_H, Wcov2, KP_H, cov_b2, out + (size_t)NH * SDIM, SDIM, nullptr, 0, 0,
       nullptr, 0, 0, NH, 16, SDIM, 3);
}
